// TreeMPNNLayer_38259568673202
// MI455X (gfx1250) — matrix operand order repaired and checked by reading, not compiled, not run on hardware
//
#include <hip/hip_runtime.h>
#include <hip/hip_bf16.h>
#include <math.h>

// ---------------------------------------------------------------------------
// TreeMPNN layer for gfx1250 (MI455X): bf16 WMMA (v_wmma_f32_16x16x32_bf16)
// for all GEMMs, f32 accumulation + f32 softmax/GRU/LayerNorm.
// Activations staged in LDS pre-packed as bf16 pairs; each 16x32 A fragment
// is two aligned ds_load_b128 per lane (branch-free).
// ---------------------------------------------------------------------------

typedef __attribute__((ext_vector_type(16))) __bf16   bf16x16;
typedef __attribute__((ext_vector_type(8)))  float    f32x8;
typedef __attribute__((ext_vector_type(8)))  unsigned uint32x8;
typedef __attribute__((ext_vector_type(4)))  unsigned uint32x4;

#define DMODEL 128
#define DEDGE  24

__device__ __forceinline__ __bf16 f2bf(float f) { return (__bf16)f; }
__device__ __forceinline__ unsigned short bf16bits(float f) {
  return __builtin_bit_cast(unsigned short, (__bf16)f);
}
// pack two floats into one dword of bf16 pair (lo in [15:0])
__device__ __forceinline__ unsigned pk2bf(float lo, float hi) {
  return ((unsigned)bf16bits(hi) << 16) | (unsigned)bf16bits(lo);
}

__device__ __forceinline__ float gelu_exact(float x) {
  return 0.5f * x * (1.0f + erff(x * 0.70710678118654752440f));
}
__device__ __forceinline__ float sigmoidf_(float x) {
  return 1.0f / (1.0f + expf(-x));
}
// order-preserving float <-> uint encoding for atomicMax on scores
__device__ __forceinline__ unsigned ord_enc(float f) {
  unsigned u = __float_as_uint(f);
  return (u & 0x80000000u) ? ~u : (u | 0x80000000u);
}
__device__ __forceinline__ float ord_dec(unsigned u) {
  unsigned v = (u & 0x80000000u) ? (u & 0x7fffffffu) : ~u;
  return __uint_as_float(v);
}

// Build a 16x32 bf16 A fragment from a packed-bf16 LDS tile.
// Per CDNA5 ISA 7.12.2 the lane's 8 dwords hold K pairs:
// dwords 0..3 -> K = kofs+0..7, dwords 4..7 -> K = kofs+16..23 (kofs = 8*grp),
// i.e. two contiguous dword groups -> two ds_load_b128.
__device__ __forceinline__ bf16x16 make_a_lds(const unsigned* __restrict__ tile,
                                              int ldw /* dwords per row */,
                                              int kbase, int lane) {
  const int m = lane & 15, grp = lane >> 4;
  const unsigned* row = tile + m * ldw + (kbase >> 1) + 4 * grp;
  uint32x4 lo = *(const uint32x4*)row;
  uint32x4 hi = *(const uint32x4*)(row + 8);
  uint32x8 d;
#pragma unroll
  for (int v = 0; v < 4; ++v) { d[v] = lo[v]; d[4 + v] = hi[v]; }
  return __builtin_bit_cast(bf16x16, d);
}

// ---------------------------------------------------------------------------
// Pack a row-major [K, Nc] f32 weight into bf16 WMMA B-fragments:
// fragment f = kt*ntiles+nt; per lane 16 contiguous bf16 (one v16bf load).
// B mapping: n = lane&15, k = kt*32 + 8*(lane>>4) + j (j<8), kt*32 + 16 + 8*(lane>>4) + (j-8) (j>=8).
// ---------------------------------------------------------------------------
__global__ void pack_weight_bf16(const float* __restrict__ W, int K, int Nc,
                                 __bf16* __restrict__ out) {
  int ktiles = (K + 31) >> 5;
  int ntiles = Nc >> 4;
  int total = ktiles * ntiles * 32;
  for (int i = blockIdx.x * blockDim.x + threadIdx.x; i < total;
       i += gridDim.x * blockDim.x) {
    int lane = i & 31;
    int f = i >> 5;
    int nt = f % ntiles, kt = f / ntiles;
    int n = nt * 16 + (lane & 15);
    int grp = lane >> 4;
    bf16x16 v;
#pragma unroll
    for (int j = 0; j < 16; ++j) {
      int k = kt * 32 + 8 * grp + ((j < 8) ? j : 8 + j);
      v[j] = f2bf((k < K) ? W[(size_t)k * Nc + n] : 0.0f);
    }
    ((bf16x16*)out)[i] = v;
  }
}

__global__ void zero_row0(float* __restrict__ sib) {
  if (threadIdx.x < DMODEL) sib[threadIdx.x] = 0.0f;
}

// ---------------------------------------------------------------------------
// Edge kernel: one wave handles 16 edges.
// ---------------------------------------------------------------------------
__global__ __launch_bounds__(32)
void edge_wmma_kernel(const float* __restrict__ h, const int* __restrict__ ei,
                      const float* __restrict__ eattr,
                      const float* __restrict__ b_msg1, const float* __restrict__ b_msg2,
                      const float* __restrict__ b_a1, const float* __restrict__ W_a2,
                      const float* __restrict__ b_a2,
                      const bf16x16* __restrict__ pk_msg1,
                      const bf16x16* __restrict__ pk_msg2,
                      const bf16x16* __restrict__ pk_a1,
                      float* __restrict__ msgs, float* __restrict__ scores,
                      unsigned* __restrict__ smax_ord, float* __restrict__ csum,
                      float* __restrict__ cnt, int E) {
  __shared__ __align__(16) unsigned s_cat1[16 * 80];   // [h_c || e || 0] 16x160 bf16
  __shared__ __align__(16) unsigned s_cat2[16 * 128];  // [h_c || h_p]   16x256 bf16
  __shared__ __align__(16) unsigned s_tw[16 * 64];     // gelu tile      16x128 bf16
  __shared__ float s_sc[16 * 32];
  __shared__ int s_ci[16];
  __shared__ int s_pi[16];

  const int lane = threadIdx.x;
  const int e0 = blockIdx.x * 16;
  if (lane < 16) {
    int e = e0 + lane;
    if (e >= E) e = E - 1;
    s_pi[lane] = ei[e];      // parent
    s_ci[lane] = ei[E + e];  // child
  }
  __syncthreads();

  // stage gathered rows: convert to packed bf16 on the fly; fold csum atomics
#pragma unroll
  for (int t = 0; t < 16; ++t) {
    int idx = lane + 32 * t;  // 0..511 float4 chunks (16 rows * 32 chunks)
    int row = idx >> 5, q = idx & 31;
    float4 vc = ((const float4*)(h + (size_t)s_ci[row] * DMODEL))[q];
    float4 vp = ((const float4*)(h + (size_t)s_pi[row] * DMODEL))[q];
    unsigned c0 = pk2bf(vc.x, vc.y), c1 = pk2bf(vc.z, vc.w);
    s_cat1[row * 80 + q * 2] = c0;
    s_cat1[row * 80 + q * 2 + 1] = c1;
    s_cat2[row * 128 + q * 2] = c0;
    s_cat2[row * 128 + q * 2 + 1] = c1;
    s_cat2[row * 128 + 64 + q * 2] = pk2bf(vp.x, vp.y);
    s_cat2[row * 128 + 64 + q * 2 + 1] = pk2bf(vp.z, vp.w);
    if (e0 + row < E) {
      float* dst = &csum[(size_t)s_pi[row] * DMODEL + q * 4];
      atomicAdd(dst + 0, vc.x);
      atomicAdd(dst + 1, vc.y);
      atomicAdd(dst + 2, vc.z);
      atomicAdd(dst + 3, vc.w);
    }
  }
#pragma unroll
  for (int t = 0; t < 3; ++t) {
    int idx = lane + 32 * t;  // 0..95 (16 rows * 6 float4)
    int row = idx / 6, q = idx % 6;
    int e = e0 + row;
    if (e >= E) e = E - 1;
    float4 va = ((const float4*)(eattr + (size_t)e * DEDGE))[q];
    s_cat1[row * 80 + 64 + q * 2] = pk2bf(va.x, va.y);
    s_cat1[row * 80 + 64 + q * 2 + 1] = pk2bf(va.z, va.w);
  }
#pragma unroll
  for (int t = 0; t < 2; ++t) {  // zero K pad 152..159 (dwords 76..79)
    int idx = lane + 32 * t;     // 0..63
    int row = idx >> 2, q = idx & 3;
    s_cat1[row * 80 + 76 + q] = 0u;
  }
  __syncthreads();

  const int col = lane & 15;
  const int hi = lane >> 4;

  // ---- message MLP layer 1: [h_c || e] @ W_msg1 + b_msg1 -> gelu -> s_tw
  {
    f32x8 acc[8];
#pragma unroll
    for (int nt = 0; nt < 8; ++nt) {
      float b = b_msg1[nt * 16 + col];
#pragma unroll
      for (int r = 0; r < 8; ++r) acc[nt][r] = b;
    }
#pragma unroll
    for (int kt = 0; kt < 5; ++kt) {
      bf16x16 a = make_a_lds(s_cat1, 80, kt * 32, lane);
#pragma unroll
      for (int nt = 0; nt < 8; ++nt) {
        bf16x16 b = pk_msg1[(kt * 8 + nt) * 32 + lane];
        acc[nt] = __builtin_amdgcn_wmma_f32_16x16x32_bf16(
            false, a, false, b, (short)0, acc[nt], false, false);
      }
    }
    unsigned short* st = (unsigned short*)s_tw;
#pragma unroll
    for (int nt = 0; nt < 8; ++nt)
#pragma unroll
      for (int r = 0; r < 8; ++r) {
        int m = r + 8 * hi;
        st[m * DMODEL + nt * 16 + col] = bf16bits(gelu_exact(acc[nt][r]));
      }
  }
  __syncthreads();

  // ---- message MLP layer 2: t @ W_msg2 + b_msg2 -> msgs (global)
  {
    f32x8 acc[8];
#pragma unroll
    for (int nt = 0; nt < 8; ++nt) {
      float b = b_msg2[nt * 16 + col];
#pragma unroll
      for (int r = 0; r < 8; ++r) acc[nt][r] = b;
    }
#pragma unroll
    for (int kt = 0; kt < 4; ++kt) {
      bf16x16 a = make_a_lds(s_tw, 64, kt * 32, lane);
#pragma unroll
      for (int nt = 0; nt < 8; ++nt) {
        bf16x16 b = pk_msg2[(kt * 8 + nt) * 32 + lane];
        acc[nt] = __builtin_amdgcn_wmma_f32_16x16x32_bf16(
            false, a, false, b, (short)0, acc[nt], false, false);
      }
    }
#pragma unroll
    for (int nt = 0; nt < 8; ++nt)
#pragma unroll
      for (int r = 0; r < 8; ++r) {
        int m = r + 8 * hi;
        int e = e0 + m;
        if (e < E) msgs[(size_t)e * DMODEL + nt * 16 + col] = acc[nt][r];
      }
  }

  // ---- attention: tanh([h_c || h_p] @ W_a1 + b_a1)
  {
    f32x8 acc[2];
#pragma unroll
    for (int nt = 0; nt < 2; ++nt) {
      float b = b_a1[nt * 16 + col];
#pragma unroll
      for (int r = 0; r < 8; ++r) acc[nt][r] = b;
    }
#pragma unroll
    for (int kt = 0; kt < 8; ++kt) {
      bf16x16 a = make_a_lds(s_cat2, 128, kt * 32, lane);
#pragma unroll
      for (int nt = 0; nt < 2; ++nt) {
        bf16x16 b = pk_a1[(kt * 2 + nt) * 32 + lane];
        acc[nt] = __builtin_amdgcn_wmma_f32_16x16x32_bf16(
            false, a, false, b, (short)0, acc[nt], false, false);
      }
    }
#pragma unroll
    for (int nt = 0; nt < 2; ++nt)
#pragma unroll
      for (int r = 0; r < 8; ++r) {
        int m = r + 8 * hi;
        s_sc[m * 32 + nt * 16 + col] = tanhf(acc[nt][r]);
      }
  }
  __syncthreads();

  if (lane < 16) {
    int e = e0 + lane;
    if (e < E) {
      float s = b_a2[0];
#pragma unroll
      for (int j = 0; j < 32; ++j) s += s_sc[lane * 32 + j] * W_a2[j];
      scores[e] = s;
      atomicMax(&smax_ord[s_pi[lane]], ord_enc(s));
      atomicAdd(&cnt[s_pi[lane]], 1.0f);
    }
  }
}

// ---------------------------------------------------------------------------
__global__ void edge_softmax_denom(const float* __restrict__ scores,
                                   const int* __restrict__ ei,
                                   const unsigned* __restrict__ smax_ord,
                                   float* __restrict__ ex,
                                   float* __restrict__ denom, int E) {
  int e = blockIdx.x * blockDim.x + threadIdx.x;
  if (e >= E) return;
  int p = ei[e];
  float x = expf(scores[e] - ord_dec(smax_ord[p]));
  ex[e] = x;
  atomicAdd(&denom[p], x);
}

__global__ void edge_scatter(const float* __restrict__ h,
                             const int* __restrict__ ei,
                             const float* __restrict__ msgs,
                             const float* __restrict__ ex,
                             const float* __restrict__ denom,
                             const float* __restrict__ csum,
                             const float* __restrict__ cnt,
                             float* __restrict__ child_agg,
                             float* __restrict__ sib, int E) {
  int idx = blockIdx.x * blockDim.x + threadIdx.x;
  int total = E * DMODEL;
  if (idx >= total) return;
  int e = idx >> 7, c = idx & 127;
  int p = ei[e], ch = ei[E + e];
  float attn = ex[e] / denom[p];
  atomicAdd(&child_agg[(size_t)p * DMODEL + c], attn * msgs[(size_t)idx]);
  float cp = cnt[p];
  float hv = h[(size_t)ch * DMODEL + c];
  float sv = (cp > 1.0f)
                 ? (csum[(size_t)p * DMODEL + c] - hv) / fmaxf(cp - 1.0f, 1.0f)
                 : 0.0f;
  sib[(size_t)ch * DMODEL + c] = sv;  // each child appears once: no conflicts
}

// ---------------------------------------------------------------------------
// Node kernel: one wave handles 16 nodes.
// ---------------------------------------------------------------------------
__global__ __launch_bounds__(32)
void node_wmma_kernel(const float* __restrict__ h,
                      const float* __restrict__ child_agg,
                      const float* __restrict__ sib,
                      const float* __restrict__ b_s1, const float* __restrict__ b_s2,
                      const float* __restrict__ b_ih, const float* __restrict__ b_hh,
                      const float* __restrict__ ln_w, const float* __restrict__ ln_b,
                      const bf16x16* __restrict__ pk_s1,
                      const bf16x16* __restrict__ pk_s2,
                      const bf16x16* __restrict__ pk_ih,
                      const bf16x16* __restrict__ pk_hh,
                      float* __restrict__ out, int Nn) {
  __shared__ float s_hf[16 * DMODEL];                 // h (f32, for gates)
  __shared__ __align__(16) unsigned s_hb[16 * 64];    // h bf16
  __shared__ __align__(16) unsigned s_cat[16 * 128];  // [ca || sf] bf16
  __shared__ __align__(16) unsigned s_sib[16 * 64];   // sib bf16
  __shared__ __align__(16) unsigned s_tmp[16 * 64];   // gelu tile bf16
  __shared__ float s_s[16 * 384];                     // gi + gh
  __shared__ float s_ghn[16 * DMODEL];                // gh_n -> h_new
  __shared__ float s_mu[16], s_rs[16];

  const int lane = threadIdx.x;
  const int n0 = blockIdx.x * 16;
  const int col = lane & 15, hi = lane >> 4;

#pragma unroll
  for (int t = 0; t < 16; ++t) {
    int idx = lane + 32 * t;
    int row = idx >> 5, q = idx & 31;
    int nr = n0 + row;
    if (nr >= Nn) nr = Nn - 1;
    size_t g = (size_t)nr * DMODEL;
    float4 vh = ((const float4*)(h + g))[q];
    float4 vc = ((const float4*)(child_agg + g))[q];
    float4 vs = ((const float4*)(sib + g))[q];
    ((float4*)(s_hf + row * DMODEL))[q] = vh;
    s_hb[row * 64 + q * 2] = pk2bf(vh.x, vh.y);
    s_hb[row * 64 + q * 2 + 1] = pk2bf(vh.z, vh.w);
    s_cat[row * 128 + q * 2] = pk2bf(vc.x, vc.y);
    s_cat[row * 128 + q * 2 + 1] = pk2bf(vc.z, vc.w);
    s_sib[row * 64 + q * 2] = pk2bf(vs.x, vs.y);
    s_sib[row * 64 + q * 2 + 1] = pk2bf(vs.z, vs.w);
  }
  __syncthreads();

  // t = gelu(sib @ W_s1 + b_s1) -> s_tmp (bf16)
  {
    f32x8 acc[8];
#pragma unroll
    for (int nt = 0; nt < 8; ++nt) {
      float b = b_s1[nt * 16 + col];
#pragma unroll
      for (int r = 0; r < 8; ++r) acc[nt][r] = b;
    }
#pragma unroll
    for (int kt = 0; kt < 4; ++kt) {
      bf16x16 a = make_a_lds(s_sib, 64, kt * 32, lane);
#pragma unroll
      for (int nt = 0; nt < 8; ++nt) {
        bf16x16 b = pk_s1[(kt * 8 + nt) * 32 + lane];
        acc[nt] = __builtin_amdgcn_wmma_f32_16x16x32_bf16(
            false, a, false, b, (short)0, acc[nt], false, false);
      }
    }
    unsigned short* st = (unsigned short*)s_tmp;
#pragma unroll
    for (int nt = 0; nt < 8; ++nt)
#pragma unroll
      for (int r = 0; r < 8; ++r)
        st[(r + 8 * hi) * DMODEL + nt * 16 + col] =
            bf16bits(gelu_exact(acc[nt][r]));
  }
  __syncthreads();

  // sf = t @ W_s2 + b_s2 -> s_cat cols 128..255 (bf16)
  {
    f32x8 acc[8];
#pragma unroll
    for (int nt = 0; nt < 8; ++nt) {
      float b = b_s2[nt * 16 + col];
#pragma unroll
      for (int r = 0; r < 8; ++r) acc[nt][r] = b;
    }
#pragma unroll
    for (int kt = 0; kt < 4; ++kt) {
      bf16x16 a = make_a_lds(s_tmp, 64, kt * 32, lane);
#pragma unroll
      for (int nt = 0; nt < 8; ++nt) {
        bf16x16 b = pk_s2[(kt * 8 + nt) * 32 + lane];
        acc[nt] = __builtin_amdgcn_wmma_f32_16x16x32_bf16(
            false, a, false, b, (short)0, acc[nt], false, false);
      }
    }
    unsigned short* st = (unsigned short*)s_cat;
#pragma unroll
    for (int nt = 0; nt < 8; ++nt)
#pragma unroll
      for (int r = 0; r < 8; ++r)
        st[(r + 8 * hi) * 256 + 128 + nt * 16 + col] = bf16bits(acc[nt][r]);
  }
  __syncthreads();

  // s = h@W_hh + [ca||sf]@W_ih + b_hh + b_ih, in 3 chunks of 8 N-tiles;
  // chunk 2 (cols 256..383) stashes the gh part separately for the n-gate.
  for (int g = 0; g < 3; ++g) {
    f32x8 acc[8];
#pragma unroll
    for (int nt = 0; nt < 8; ++nt) {
      float b = b_hh[(g * 8 + nt) * 16 + col];
#pragma unroll
      for (int r = 0; r < 8; ++r) acc[nt][r] = b;
    }
#pragma unroll
    for (int kt = 0; kt < 4; ++kt) {
      bf16x16 a = make_a_lds(s_hb, 64, kt * 32, lane);
#pragma unroll
      for (int nt = 0; nt < 8; ++nt) {
        bf16x16 b = pk_hh[(kt * 24 + g * 8 + nt) * 32 + lane];
        acc[nt] = __builtin_amdgcn_wmma_f32_16x16x32_bf16(
            false, a, false, b, (short)0, acc[nt], false, false);
      }
    }
    if (g == 2) {
#pragma unroll
      for (int nt = 0; nt < 8; ++nt)
#pragma unroll
        for (int r = 0; r < 8; ++r)
          s_ghn[(r + 8 * hi) * DMODEL + nt * 16 + col] = acc[nt][r];
    }
#pragma unroll
    for (int nt = 0; nt < 8; ++nt) {
      float b = b_ih[(g * 8 + nt) * 16 + col];
#pragma unroll
      for (int r = 0; r < 8; ++r) acc[nt][r] += b;
    }
#pragma unroll
    for (int kt = 0; kt < 8; ++kt) {
      bf16x16 a = make_a_lds(s_cat, 128, kt * 32, lane);
#pragma unroll
      for (int nt = 0; nt < 8; ++nt) {
        bf16x16 b = pk_ih[(kt * 24 + g * 8 + nt) * 32 + lane];
        acc[nt] = __builtin_amdgcn_wmma_f32_16x16x32_bf16(
            false, a, false, b, (short)0, acc[nt], false, false);
      }
    }
#pragma unroll
    for (int nt = 0; nt < 8; ++nt)
#pragma unroll
      for (int r = 0; r < 8; ++r)
        s_s[(r + 8 * hi) * 384 + (g * 8 + nt) * 16 + col] = acc[nt][r];
  }
  __syncthreads();

  // GRU gates: n = tanh(s_n + (r-1)*gh_n); h_new overwrites s_ghn
#pragma unroll
  for (int t = 0; t < 64; ++t) {
    int idx = lane + 32 * t;
    int m = idx >> 7, c = idx & 127;
    float r = sigmoidf_(s_s[m * 384 + c]);
    float z = sigmoidf_(s_s[m * 384 + 128 + c]);
    float nv = tanhf(s_s[m * 384 + 256 + c] + (r - 1.0f) * s_ghn[m * DMODEL + c]);
    s_ghn[m * DMODEL + c] = (1.0f - z) * nv + z * s_hf[m * DMODEL + c];
  }
  __syncthreads();

  if (lane < 16) {
    float mu = 0.0f;
    for (int c = 0; c < DMODEL; ++c) mu += s_ghn[lane * DMODEL + c];
    mu *= (1.0f / DMODEL);
    float var = 0.0f;
    for (int c = 0; c < DMODEL; ++c) {
      float d = s_ghn[lane * DMODEL + c] - mu;
      var += d * d;
    }
    var *= (1.0f / DMODEL);
    s_mu[lane] = mu;
    s_rs[lane] = rsqrtf(var + 1e-5f);
  }
  __syncthreads();

#pragma unroll
  for (int t = 0; t < 64; ++t) {
    int idx = lane + 32 * t;
    int m = idx >> 7, c = idx & 127;
    if (n0 + m < Nn) {
      float v = (s_ghn[m * DMODEL + c] - s_mu[m]) * s_rs[m];
      out[(size_t)(n0 + m) * DMODEL + c] = v * ln_w[c] + ln_b[c];
    }
  }
}

// ---------------------------------------------------------------------------
extern "C" void kernel_launch(void* const* d_in, const int* in_sizes, int n_in,
                              void* d_out, int out_size, void* d_ws,
                              size_t ws_size, hipStream_t stream) {
  const float* h = (const float*)d_in[0];
  const int* ei = (const int*)d_in[1];  // [2, E] int32
  const float* eattr = (const float*)d_in[2];
  const float* W_msg1 = (const float*)d_in[3];
  const float* b_msg1 = (const float*)d_in[4];
  const float* W_msg2 = (const float*)d_in[5];
  const float* b_msg2 = (const float*)d_in[6];
  const float* W_a1 = (const float*)d_in[7];
  const float* b_a1 = (const float*)d_in[8];
  const float* W_a2 = (const float*)d_in[9];
  const float* b_a2 = (const float*)d_in[10];
  const float* W_s1 = (const float*)d_in[11];
  const float* b_s1 = (const float*)d_in[12];
  const float* W_s2 = (const float*)d_in[13];
  const float* b_s2 = (const float*)d_in[14];
  const float* W_ih = (const float*)d_in[15];
  const float* W_hh = (const float*)d_in[16];
  const float* b_ih = (const float*)d_in[17];
  const float* b_hh = (const float*)d_in[18];
  const float* ln_w = (const float*)d_in[19];
  const float* ln_b = (const float*)d_in[20];

  const int Nn = in_sizes[0] / DMODEL;
  const int E = in_sizes[1] / 2;

  char* ws = (char*)d_ws;
  auto alignup = [](size_t x) { return (x + 255) & ~(size_t)255; };
  size_t off = 0;
  // --- zero block (reset every call; harness does not re-poison) ---
  unsigned* smax = (unsigned*)(ws + off); off = alignup(off + (size_t)Nn * 4);
  float* denom = (float*)(ws + off);      off = alignup(off + (size_t)Nn * 4);
  float* cnt = (float*)(ws + off);        off = alignup(off + (size_t)Nn * 4);
  float* csum = (float*)(ws + off);       off = alignup(off + (size_t)Nn * DMODEL * 4);
  float* cagg = (float*)(ws + off);       off = alignup(off + (size_t)Nn * DMODEL * 4);
  size_t zero_bytes = off;
  // --- fully-overwritten scratch ---
  float* sib = (float*)(ws + off);    off = alignup(off + (size_t)Nn * DMODEL * 4);
  float* msgs = (float*)(ws + off);   off = alignup(off + (size_t)E * DMODEL * 4);
  float* scores = (float*)(ws + off); off = alignup(off + (size_t)E * 4);
  float* ex = (float*)(ws + off);     off = alignup(off + (size_t)E * 4);
  __bf16* pk = (__bf16*)(ws + off);
  __bf16* pk_msg1 = pk + (size_t)0 * 512;    // 5*8  = 40 frags
  __bf16* pk_msg2 = pk + (size_t)40 * 512;   // 4*8  = 32
  __bf16* pk_a1 = pk + (size_t)72 * 512;     // 8*2  = 16
  __bf16* pk_s1 = pk + (size_t)88 * 512;     // 4*8  = 32
  __bf16* pk_s2 = pk + (size_t)120 * 512;    // 4*8  = 32
  __bf16* pk_ih = pk + (size_t)152 * 512;    // 8*24 = 192
  __bf16* pk_hh = pk + (size_t)344 * 512;    // 4*24 = 96

  hipMemsetAsync(ws, 0, zero_bytes, stream);
  zero_row0<<<1, 128, 0, stream>>>(sib);  // node 0 has no parent edge

  pack_weight_bf16<<<32, 256, 0, stream>>>(W_msg1, DMODEL + DEDGE, DMODEL, pk_msg1);
  pack_weight_bf16<<<32, 256, 0, stream>>>(W_msg2, DMODEL, DMODEL, pk_msg2);
  pack_weight_bf16<<<32, 256, 0, stream>>>(W_a1, 2 * DMODEL, 32, pk_a1);
  pack_weight_bf16<<<32, 256, 0, stream>>>(W_s1, DMODEL, DMODEL, pk_s1);
  pack_weight_bf16<<<32, 256, 0, stream>>>(W_s2, DMODEL, DMODEL, pk_s2);
  pack_weight_bf16<<<32, 256, 0, stream>>>(W_ih, 2 * DMODEL, 3 * DMODEL, pk_ih);
  pack_weight_bf16<<<32, 256, 0, stream>>>(W_hh, DMODEL, 3 * DMODEL, pk_hh);

  int etiles = (E + 15) / 16;
  edge_wmma_kernel<<<etiles, 32, 0, stream>>>(
      h, ei, eattr, b_msg1, b_msg2, b_a1, W_a2, b_a2, (const bf16x16*)pk_msg1,
      (const bf16x16*)pk_msg2, (const bf16x16*)pk_a1, msgs, scores, smax, csum,
      cnt, E);

  edge_softmax_denom<<<(E + 255) / 256, 256, 0, stream>>>(scores, ei, smax, ex,
                                                          denom, E);

  int total = E * DMODEL;
  edge_scatter<<<(total + 255) / 256, 256, 0, stream>>>(
      h, ei, msgs, ex, denom, csum, cnt, cagg, sib, E);

  int ntiles = (Nn + 15) / 16;
  node_wmma_kernel<<<ntiles, 32, 0, stream>>>(
      h, cagg, sib, b_s1, b_s2, b_ih, b_hh, ln_w, ln_b, (const bf16x16*)pk_s1,
      (const bf16x16*)pk_s2, (const bf16x16*)pk_ih, (const bf16x16*)pk_hh,
      (float*)d_out, Nn);
}
